// Model_26044681683416
// MI455X (gfx1250) — matrix operand order repaired and checked by reading, not compiled, not run on hardware
//
#include <hip/hip_runtime.h>
#include <hip/hip_bf16.h>

// MI455X / gfx1250, wave32.
// Pipeline:
//   k_packW  : pre-pack W_att into WMMA B-operand f16 lane layout (once, tiny)
//   k_proj   : h = x @ W_att        (WMMA f16, unguarded b128 streams)
//   k_scores : GAT score vectors + exp factor tables (exp-free O(B^2) later)
//   k_gat    : GAT softmax-aggregate, LDS tiles filled by async-to-LDS copies
//   k_sattn  : rownorm(sigmoid(x3 x3^T)) @ x3, flash-style, WMMA + double-
//              buffered async-to-LDS j-tiles
//   k_head   : maxpool + fc + log_softmax

typedef __attribute__((ext_vector_type(16))) _Float16 v16h;
typedef __attribute__((ext_vector_type(8)))  float    v8f;

#define B_NODES 4096
#define KDIM    784
#define FEAT    32

__device__ __forceinline__ v8f wmma16(v16h a, v16h b, v8f c) {
    // D = A(16x32 f16) x B(32x16 f16) + C(16x16 f32)
    return __builtin_amdgcn_wmma_f32_16x16x32_f16(
        false, a, false, b, (short)0, c, false, false);
}

__device__ __forceinline__ void pack8(v16h& a, int base, float4 q0, float4 q1) {
    a[base + 0] = (_Float16)q0.x; a[base + 1] = (_Float16)q0.y;
    a[base + 2] = (_Float16)q0.z; a[base + 3] = (_Float16)q0.w;
    a[base + 4] = (_Float16)q1.x; a[base + 5] = (_Float16)q1.y;
    a[base + 6] = (_Float16)q1.z; a[base + 7] = (_Float16)q1.w;
}

// LDS offset of a __shared__ object: generic pointer low 32 bits are the
// group-segment byte offset (LDS aperture: addr[63:32]=SHARED_BASE).
__device__ __forceinline__ unsigned lds_off(const void* p) {
    return (unsigned)(uintptr_t)p;
}

// Async copy 32B (global -> LDS), tracked by ASYNCcnt.
__device__ __forceinline__ void async_cp32(unsigned lds, const float* g) {
    asm volatile("global_load_async_to_lds_b128 %0, %1, off\n\t"
                 "global_load_async_to_lds_b128 %0, %1, off offset:16"
                 :: "v"(lds), "v"(g) : "memory");
}
// Async copy 64B (global -> LDS).
__device__ __forceinline__ void async_cp64(unsigned lds, const float* g) {
    asm volatile("global_load_async_to_lds_b128 %0, %1, off\n\t"
                 "global_load_async_to_lds_b128 %0, %1, off offset:16\n\t"
                 "global_load_async_to_lds_b128 %0, %1, off offset:32\n\t"
                 "global_load_async_to_lds_b128 %0, %1, off offset:48"
                 :: "v"(lds), "v"(g) : "memory");
}
__device__ __forceinline__ void async_wait0() {
    asm volatile("s_wait_asynccnt 0" ::: "memory");
}

// ---------------------------------------------------------------------------
// Kernel 0: pack W_att[784,32] f32 -> WMMA B-operand f16 layout.
// Layout: [nt(2)][ks(25)][lane(32)][e(16)] halves; K-tail clamped (the A
// operand zeroes K>=784, so clamped-but-finite B values contribute 0).
// ---------------------------------------------------------------------------
__global__ void __launch_bounds__(256) k_packW(const float* __restrict__ Watt,
                                               _Float16* __restrict__ wpk) {
    const int t = blockIdx.x * blockDim.x + threadIdx.x;
    if (t >= 2 * 25 * 32) return;
    const int lane = t & 31;
    const int ks = (t >> 5) % 25;
    const int nt = t / (25 * 32);
    const int m = lane & 15, hi = lane >> 4;
    const int k0 = ks * 32;
    #pragma unroll
    for (int e = 0; e < 16; ++e) {
        int kg = k0 + ((e < 8) ? (8 * hi + e) : (16 + 8 * hi + (e - 8)));
        kg = (kg < KDIM) ? kg : (KDIM - 1);
        wpk[(size_t)t * 16 + e] = (_Float16)Watt[(size_t)kg * FEAT + nt * 16 + m];
    }
}

// ---------------------------------------------------------------------------
// Kernel 1: h = x1[4096,784] @ W_att[784,32].  One wave per 16x32 tile
// (A reused across both N-halves). 256 waves; memory bound on x (12.8 MB).
// ---------------------------------------------------------------------------
__global__ void __launch_bounds__(128) k_proj(const float* __restrict__ x,
                                              const _Float16* __restrict__ wpk,
                                              float* __restrict__ h_) {
    const int wid  = blockIdx.x * 4 + (threadIdx.x >> 5);   // M-tile 0..255
    const int lane = threadIdx.x & 31;
    const int i0 = wid * 16;
    const int m = lane & 15, hi = lane >> 4, kk = hi * 8;

    const float*    arow = x + (size_t)(i0 + m) * KDIM;
    const _Float16* wp0  = wpk + (size_t)lane * 16;

    v8f c0 = {}, c1 = {};

    for (int ks = 0; ks < 24; ++ks) {         // full K=32 steps
        const int k0 = ks * 32;
        v16h a;
        {
            float4 q0 = *(const float4*)(arow + k0 + kk);
            float4 q1 = *(const float4*)(arow + k0 + kk + 4);
            float4 q2 = *(const float4*)(arow + k0 + 16 + kk);
            float4 q3 = *(const float4*)(arow + k0 + 16 + kk + 4);
            pack8(a, 0, q0, q1);
            pack8(a, 8, q2, q3);
        }
        v16h b0 = *(const v16h*)(wp0 + (size_t)ks * 512);        // nt=0
        v16h b1 = *(const v16h*)(wp0 + (size_t)(25 + ks) * 512); // nt=1
        c0 = wmma16(a, b0, c0);
        c1 = wmma16(a, b1, c1);
    }
    {   // K tail: 768..783 (upper half of A zeroed; B was clamp-packed)
        const int k0 = 768;
        v16h a;
        float4 q0 = *(const float4*)(arow + k0 + kk);
        float4 q1 = *(const float4*)(arow + k0 + kk + 4);
        pack8(a, 0, q0, q1);
        #pragma unroll
        for (int e = 8; e < 16; ++e) a[e] = (_Float16)0.f;
        v16h b0 = *(const v16h*)(wp0 + (size_t)24 * 512);
        v16h b1 = *(const v16h*)(wp0 + (size_t)49 * 512);
        c0 = wmma16(a, b0, c0);
        c1 = wmma16(a, b1, c1);
    }
    #pragma unroll
    for (int v = 0; v < 8; ++v) {
        const size_t row = (size_t)(i0 + v + 8 * hi) * FEAT;
        h_[row + m]      = c0[v];
        h_[row + 16 + m] = c1[v];
    }
}

// ---------------------------------------------------------------------------
// Kernel 2: per-(node,head) scores + exp factor tables.
// exp(lrelu(s+t)) = (s+t>0) ? e^s e^t : e^{0.2s} e^{0.2t}
// ---------------------------------------------------------------------------
__global__ void __launch_bounds__(256) k_scores(const float* __restrict__ h_,
                                                const float* __restrict__ a_src,
                                                const float* __restrict__ a_dst,
                                                float* __restrict__ ss, float* __restrict__ sd,
                                                float* __restrict__ Es, float* __restrict__ Fs,
                                                float* __restrict__ Et, float* __restrict__ Ft) {
    const int i = blockIdx.x * blockDim.x + threadIdx.x;   // 4096
    const float* hr = h_ + (size_t)i * FEAT;
    #pragma unroll
    for (int hh = 0; hh < 4; ++hh) {
        float s = 0.f, t = 0.f;
        #pragma unroll
        for (int d = 0; d < 8; ++d) {
            const float hv = hr[hh * 8 + d];
            s += hv * a_src[hh * 8 + d];
            t += hv * a_dst[hh * 8 + d];
        }
        const int idx = i * 4 + hh;
        ss[idx] = s;            sd[idx] = t;
        Es[idx] = __expf(s);    Fs[idx] = __expf(0.2f * s);
        Et[idx] = __expf(t);    Ft[idx] = __expf(0.2f * t);
    }
}

// ---------------------------------------------------------------------------
// Kernel 3: GAT aggregation (exp-free inner loop), fused BN + maxpool.
// Block = 64 nodes x 4 heads; 128-row j tiles staged with async-to-LDS.
// ---------------------------------------------------------------------------
__global__ void __launch_bounds__(256) k_gat(const float* __restrict__ h_,
                                             const float* __restrict__ ss,
                                             const float* __restrict__ Es, const float* __restrict__ Fs,
                                             const float* __restrict__ sd,
                                             const float* __restrict__ Et, const float* __restrict__ Ft,
                                             const float* __restrict__ gamma,
                                             const float* __restrict__ beta,
                                             float* __restrict__ x3, float* __restrict__ x4) {
    __shared__ float h_sh[128 * FEAT];
    __shared__ float t_sh[128 * 4];
    __shared__ float et_sh[128 * 4];
    __shared__ float ft_sh[128 * 4];

    const int tid = threadIdx.x;
    const int il = tid >> 2, hh = tid & 3;
    const int i = blockIdx.x * 64 + il;
    const float s_i = ss[i * 4 + hh];
    const float es_i = Es[i * 4 + hh], fs_i = Fs[i * 4 + hh];

    const unsigned hsh_lds = lds_off(&h_sh[tid * 16]);

    float accw = 0.f, acc[8] = {};

    for (int jc = 0; jc < B_NODES / 128; ++jc) {
        const int j0 = jc * 128;
        __syncthreads();                       // tile buffers free to overwrite
        async_cp64(hsh_lds, h_ + (size_t)j0 * FEAT + tid * 16);
        #pragma unroll
        for (int r = 0; r < 2; ++r) {
            const int idx = tid + r * 256;
            t_sh[idx]  = sd[j0 * 4 + idx];
            et_sh[idx] = Et[j0 * 4 + idx];
            ft_sh[idx] = Ft[j0 * 4 + idx];
        }
        async_wait0();
        __syncthreads();

        for (int j = 0; j < 128; ++j) {
            const float t = t_sh[j * 4 + hh];
            const float wgt = (s_i + t > 0.f) ? es_i * et_sh[j * 4 + hh]
                                              : fs_i * ft_sh[j * 4 + hh];
            accw += wgt;
            const float* hj = &h_sh[j * FEAT + hh * 8];
            #pragma unroll
            for (int d = 0; d < 8; ++d) acc[d] = fmaf(wgt, hj[d], acc[d]);
        }
    }

    const float invw = 1.f / accw;
    const float bninv = 0.99999500003749968f;  // 1/sqrt(1 + 1e-5)
    float x2v[8];
    #pragma unroll
    for (int d = 0; d < 8; ++d) x2v[d] = acc[d] * invw;
    #pragma unroll
    for (int d = 0; d < 8; ++d)
        x3[(size_t)i * FEAT + hh * 8 + d] =
            gamma[hh * 8 + d] * x2v[d] * bninv + beta[hh * 8 + d];
    #pragma unroll
    for (int p = 0; p < 4; ++p)
        x4[(size_t)i * 16 + hh * 4 + p] = fmaxf(x2v[2 * p], x2v[2 * p + 1]);
}

// ---------------------------------------------------------------------------
// Kernel 4: x5 = rownorm(sigmoid(x3 x3^T)) @ x3, flash-style (never builds
// the 4096x4096 matrix). WMMA both matmuls; j-tiles double-buffered via
// async-to-LDS; P re-layout via per-wave LDS slice.
// ---------------------------------------------------------------------------
__global__ void __launch_bounds__(128) k_sattn(const float* __restrict__ x3,
                                               float* __restrict__ x5) {
    __shared__ float xj_sh[2][32 * FEAT];     // double-buffered j-tile (8 KB)
    __shared__ float p_sh[4][16 * 32];        // per-wave sigmoid tile (8 KB)

    const int tid = threadIdx.x;
    const int w = tid >> 5, lane = tid & 31;
    const int m = lane & 15, hi = lane >> 4, kk = hi * 8;
    const int i0 = blockIdx.x * 64 + w * 16;
    const int NT = B_NODES / 32;

    // Xi A-operand (16x32 f16), in registers for the whole kernel.
    v16h aXi;
    {
        const float* xr = x3 + (size_t)(i0 + m) * FEAT;
        float4 q0 = *(const float4*)(xr + kk);
        float4 q1 = *(const float4*)(xr + kk + 4);
        float4 q2 = *(const float4*)(xr + 16 + kk);
        float4 q3 = *(const float4*)(xr + 16 + kk + 4);
        pack8(aXi, 0, q0, q1);
        pack8(aXi, 8, q2, q3);
    }

    const unsigned xj_lds0 = lds_off(&xj_sh[0][tid * 8]);
    const unsigned xj_lds1 = lds_off(&xj_sh[1][tid * 8]);

    v8f acc0 = {}, acc1 = {};
    float dsum[8] = {};

    async_cp32(xj_lds0, x3 + (size_t)tid * 8);   // prologue: tile 0

    for (int jt = 0; jt < NT; ++jt) {
        async_wait0();
        __syncthreads();      // current tile resident; prev-prev buffer idle
        if (jt + 1 < NT)
            async_cp32((jt & 1) ? xj_lds0 : xj_lds1,
                       x3 + (size_t)(jt + 1) * 32 * FEAT + tid * 8);
        const float* xj = &xj_sh[jt & 1][0];

        // S = Xi @ Xj^T (two 16-wide j halves), sigmoid, stash to LDS.
        #pragma unroll
        for (int jj = 0; jj < 2; ++jj) {
            const float* brow = &xj[(jj * 16 + m) * FEAT + 8 * hi];
            v16h b;
            {
                float4 q0 = *(const float4*)(brow);
                float4 q1 = *(const float4*)(brow + 4);
                float4 q2 = *(const float4*)(brow + 16);
                float4 q3 = *(const float4*)(brow + 20);
                pack8(b, 0, q0, q1);
                pack8(b, 8, q2, q3);
            }
            v8f cz = {};
            v8f s = wmma16(aXi, b, cz);
            #pragma unroll
            for (int v = 0; v < 8; ++v) {
                const float sig = 1.f / (1.f + __expf(-s[v]));
                dsum[v] += sig;
                p_sh[w][(v + 8 * hi) * 32 + jj * 16 + m] = sig;
            }
        }
        // (same-wave LDS ops are in-order; DScnt waits inserted by compiler)

        // P A-operand (16 rows x 32 j, f16) from LDS.
        v16h a2;
        {
            const float* pr = &p_sh[w][m * 32 + kk];
            float4 q0 = *(const float4*)pr;
            float4 q1 = *(const float4*)(pr + 4);
            const float* pr2 = &p_sh[w][m * 32 + 16 + kk];
            float4 q2 = *(const float4*)pr2;
            float4 q3 = *(const float4*)(pr2 + 4);
            pack8(a2, 0, q0, q1);
            pack8(a2, 8, q2, q3);
        }
        // acc += P @ Xj (two 16-wide feature halves)
        #pragma unroll
        for (int nt = 0; nt < 2; ++nt) {
            v16h b2;
            #pragma unroll
            for (int e = 0; e < 16; ++e)
                b2[e] = (_Float16)xj[((e < 8) ? (8 * hi + e) : (16 + 8 * hi + (e - 8))) * FEAT + nt * 16 + m];
            if (nt == 0) acc0 = wmma16(a2, b2, acc0);
            else         acc1 = wmma16(a2, b2, acc1);
        }
    }

    // Row-sum denominators: reduce across the 16 lanes of each half-wave.
    #pragma unroll
    for (int v = 0; v < 8; ++v) {
        float d = dsum[v];
        d += __shfl_xor(d, 1, 32);
        d += __shfl_xor(d, 2, 32);
        d += __shfl_xor(d, 4, 32);
        d += __shfl_xor(d, 8, 32);
        dsum[v] = d;
    }
    #pragma unroll
    for (int v = 0; v < 8; ++v) {
        const float inv = 1.f / dsum[v];
        const size_t row = (size_t)(i0 + v + 8 * hi) * FEAT;
        x5[row + m]      = acc0[v] * inv;
        x5[row + 16 + m] = acc1[v] * inv;
    }
}

// ---------------------------------------------------------------------------
// Kernel 5: x6 = maxpool(x5) + x4; logits = x6 @ fc_w + fc_b; log_softmax.
// ---------------------------------------------------------------------------
__global__ void __launch_bounds__(256) k_head(const float* __restrict__ x5,
                                              const float* __restrict__ x4,
                                              const float* __restrict__ fc_w,
                                              const float* __restrict__ fc_b,
                                              float* __restrict__ out) {
    const int i = blockIdx.x * blockDim.x + threadIdx.x;   // 4096
    const float* r5 = x5 + (size_t)i * FEAT;
    const float* r4 = x4 + (size_t)i * 16;
    float x6[16];
    #pragma unroll
    for (int f = 0; f < 16; ++f)
        x6[f] = fmaxf(r5[2 * f], r5[2 * f + 1]) + r4[f];
    float lg[10];
    #pragma unroll
    for (int o = 0; o < 10; ++o) {
        float a = fc_b[o];
        #pragma unroll
        for (int f = 0; f < 16; ++f) a = fmaf(x6[f], fc_w[f * 10 + o], a);
        lg[o] = a;
    }
    float mx = lg[0];
    #pragma unroll
    for (int o = 1; o < 10; ++o) mx = fmaxf(mx, lg[o]);
    float se = 0.f;
    #pragma unroll
    for (int o = 0; o < 10; ++o) se += __expf(lg[o] - mx);
    const float lse = mx + __logf(se);
    #pragma unroll
    for (int o = 0; o < 10; ++o) out[(size_t)i * 10 + o] = lg[o] - lse;
}

// ---------------------------------------------------------------------------
extern "C" void kernel_launch(void* const* d_in, const int* in_sizes, int n_in,
                              void* d_out, int out_size, void* d_ws, size_t ws_size,
                              hipStream_t stream) {
    const float* x     = (const float*)d_in[0];
    const float* Watt  = (const float*)d_in[1];
    const float* a_src = (const float*)d_in[2];
    const float* a_dst = (const float*)d_in[3];
    const float* gamma = (const float*)d_in[4];
    const float* beta  = (const float*)d_in[5];
    const float* fc_w  = (const float*)d_in[6];
    const float* fc_b  = (const float*)d_in[7];
    // d_in[8] = train flag, eval path only.

    float* ws = (float*)d_ws;
    float* h_ = ws;                 // 4096*32
    float* ss = h_ + 131072;        // 4096*4
    float* sd = ss + 16384;
    float* Es = sd + 16384;
    float* Fs = Es + 16384;
    float* Et = Fs + 16384;
    float* Ft = Et + 16384;
    float* x3 = Ft + 16384;         // 4096*32
    float* x4 = x3 + 131072;        // 4096*16
    float* x5 = x4 + 65536;         // 4096*32
    _Float16* wpk = (_Float16*)(x5 + 131072);   // 2*25*32*16 halves (50 KB)

    k_packW <<<7,   256, 0, stream>>>(Watt, wpk);
    k_proj  <<<64,  128, 0, stream>>>(x, wpk, h_);
    k_scores<<<16,  256, 0, stream>>>(h_, a_src, a_dst, ss, sd, Es, Fs, Et, Ft);
    k_gat   <<<64,  256, 0, stream>>>(h_, ss, Es, Fs, sd, Et, Ft, gamma, beta, x3, x4);
    k_sattn <<<64,  128, 0, stream>>>(x3, x5);
    k_head  <<<16,  256, 0, stream>>>(x5, x4, fc_w, fc_b, (float*)d_out);
}
